// MultiHeadAttention_37065567765242
// MI455X (gfx1250) — matrix operand order repaired and checked by reading, not compiled, not run on hardware
//
#include <hip/hip_runtime.h>
#include <hip/hip_bf16.h>

// ---------------------------------------------------------------------------
// MHA forward for gfx1250 (CDNA5): bf16 WMMA (16x16x32), fp32 accumulate.
//   Weights          : pre-converted fp32 -> bf16 once (v_cvt_pk_bf16_f32).
//   Q projection     : row-major, pre-scaled by log2(e)/sqrt(64)
//   K projection     : rows permuted within 32-token groups ((s&1)<<4 | s>>1)
//                      so attention logit columns come out in kv-pair order.
//   V projection     : written transposed as Vt[b, d, s] (s contiguous).
//   Flash attention  : K/V tiles staged by the Tensor Data Mover
//                      (tensor_load_to_lds + TENSORcnt) — zero VGPR staging;
//                      bf16 QK^T + online softmax + ones-column denominator
//                      WMMA + bf16 P*V, double-buffered LDS tiles.
//   Output projection: ctx @ Wo^T + bo   (bf16 -> fp32)
// Scratch (d_ws): Qp, Kp, Vt, ctx (8MB each) + bf16 weights (8MB) = 40MB.
// ---------------------------------------------------------------------------

typedef __bf16 v16bf __attribute__((ext_vector_type(16)));
typedef float  v8f   __attribute__((ext_vector_type(8)));
typedef unsigned int   u32;
typedef unsigned short u16;
typedef u32 v4u __attribute__((ext_vector_type(4)));
typedef int v8i __attribute__((ext_vector_type(8)));
typedef int v4i __attribute__((ext_vector_type(4)));

static constexpr int Bb = 2, Ss = 2048, Dd = 1024, Hh = 16, DEPTH = 64;

__device__ __forceinline__ u16 f2bf(float f) {
  union { float f; u32 u; } x; x.f = f;
  u32 u = x.u;
  u += 0x7fffu + ((u >> 16) & 1u);        // round-to-nearest-even
  return (u16)(u >> 16);
}

// two f32 -> packed bf16 in one VALU op
__device__ __forceinline__ u32 cvt_pk_bf16(float lo, float hi) {
  u32 d;
  asm("v_cvt_pk_bf16_f32 %0, %1, %2" : "=v"(d) : "v"(lo), "v"(hi));
  return d;
}

__device__ __forceinline__ v8f wmma_bf16(v16bf a, v16bf b, v8f c) {
  return __builtin_amdgcn_wmma_f32_16x16x32_bf16(
      false, a, false, b, (short)0, c, false, false);
}

// bare max (WMMA outputs are never sNaN; skip fmaxf canonicalization)
__device__ __forceinline__ float vmax(float a, float b) {
  float d;
  asm("v_max_num_f32 %0, %1, %2" : "=v"(d) : "v"(a), "v"(b));
  return d;
}

// xor-shuffle via ds_swizzle (group-of-32 mode: and=0x1f, xor=MASK)
template <int IMM>
__device__ __forceinline__ float swzx(float v) {
  int r = __builtin_amdgcn_ds_swizzle(__builtin_bit_cast(int, v), IMM);
  return __builtin_bit_cast(float, r);
}
#define SWZ_X1 0x041f
#define SWZ_X2 0x081f
#define SWZ_X4 0x101f
#define SWZ_X8 0x201f

union Frag {
  struct { uint4 lo; uint4 hi; } u;   // 32 bytes
  v16bf v;
};

// ---- TDM: copy a 2D tile (DIM1 rows of DIM0 bf16, row stride STRIDE0
//      elements) from global to LDS, padding rows per the D# pad fields. ----
template <int DIM0, int DIM1, int STRIDE0, int PI, int PA>
__device__ __forceinline__ void tdm_tile_to_lds(const u16* g, u32 lds_off) {
  unsigned long long ga = (unsigned long long)(__UINTPTR_TYPE__)g;
  v4u g0 = { 1u,                                   // count=1, user desc
             lds_off,                              // LDS byte address
             (u32)ga,                              // global_addr[31:0]
             ((u32)(ga >> 32) & 0x01ffffffu) | (2u << 30) };  // addr hi | type=2
  v8i g1;
  g1[0] = (1 << 16) | (1 << 20) | (PI << 22) | (PA << 25);
  g1[1] = (DIM0 & 0xffff) << 16;                   // tensor_dim0[15:0]
  g1[2] = (DIM0 >> 16) | ((DIM1 & 0xffff) << 16);  // dim0 hi | tensor_dim1 lo
  g1[3] = (DIM1 >> 16) | ((DIM0 & 0xffff) << 16);  // dim1 hi | tile_dim0
  g1[4] = DIM1;                                    // tile_dim1 (tile_dim2=0)
  g1[5] = STRIDE0;                                 // tensor_dim0_stride lo
  g1[6] = 0;
  g1[7] = 0;
  v4i z4 = { 0, 0, 0, 0 };
#if defined(__clang_major__) && __clang_major__ >= 23
  v8i z8 = { 0, 0, 0, 0, 0, 0, 0, 0 };
  __builtin_amdgcn_tensor_load_to_lds(g0, g1, z4, z4, z8, 0);
#else
  __builtin_amdgcn_tensor_load_to_lds(g0, g1, z4, z4, 0);
#endif
}

// -------- packed 4-element loaders (to 4 bf16 in a uint2) --------------------
template <typename T> struct LdPk;
template <> struct LdPk<float> {
  static __device__ __forceinline__ uint2 load(const float* p) {
    float4 v = *(const float4*)p;
    uint2 r;
    r.x = cvt_pk_bf16(v.x, v.y);
    r.y = cvt_pk_bf16(v.z, v.w);
    return r;
  }
};
template <> struct LdPk<u16> {
  static __device__ __forceinline__ uint2 load(const u16* p) {
    return *(const uint2*)p;
  }
};

template <typename T> __device__ __forceinline__ void stout(T* p, float v);
template <> __device__ __forceinline__ void stout<float>(float* p, float v) { *p = v; }
template <> __device__ __forceinline__ void stout<u16>(u16* p, float v)     { *p = f2bf(v); }

// ---------------------------------------------------------------------------
// One-shot weight conversion: fp32 [N*K] -> bf16, vectorized float4 -> uint2.
// ---------------------------------------------------------------------------
__global__ __launch_bounds__(256) void cvt_bf16_kernel(
    const float* __restrict__ src, u16* __restrict__ dst, int nQuads) {
  int i = blockIdx.x * 256 + threadIdx.x;
  if (i < nQuads) ((uint2*)dst)[i] = LdPk<float>::load(src + i * 4);
}

// ---------------------------------------------------------------------------
// GEMM: C = (A[M,K] * Wb[N,K]^T + bias) * outScale   (Wb already bf16)
// LAYOUT 0: row-major.  1: transposed [b, n, s].  2: row-major with token
// rows permuted within 32-row groups ((s&1)<<4 | s>>1) for the K operand.
// ---------------------------------------------------------------------------
template <typename InT, typename OutT, int LAYOUT>
__global__ __launch_bounds__(256) void gemm_bias_bf16(
    const InT* __restrict__ A, const u16* __restrict__ Wb,
    const float* __restrict__ bias, OutT* __restrict__ C,
    int M, int N, int K, float outScale) {
  constexpr int BM = 128, BN = 128, BK = 32;
  constexpr int LDA = 40;                   // u16 stride (20 dwords, padded)
  __shared__ u16 aS[BM * LDA];
  __shared__ u16 wS[BN * LDA];

  const int tid  = threadIdx.x;
  const int lane = tid & 31;
  const int wv   = tid >> 5;
  const int half = lane >> 4, ln = lane & 15;
  const int wm   = wv & 3,    wn = wv >> 2;         // 4 waves in M, 2 in N
  const int m0   = blockIdx.y * BM, n0 = blockIdx.x * BN;

  v8f c[2][4];
#pragma unroll
  for (int mt = 0; mt < 2; ++mt)
#pragma unroll
    for (int nt = 0; nt < 4; ++nt)
#pragma unroll
      for (int r = 0; r < 8; ++r) c[mt][nt][r] = 0.0f;

  for (int k0 = 0; k0 < K; k0 += BK) {
    __syncthreads();
#pragma unroll
    for (int i = 0; i < 4; ++i) {
      int idx = tid + i * 256;              // 1024 quads
      int r   = idx >> 3;
      int cc  = (idx & 7) * 4;
      *(uint2*)(aS + r * LDA + cc) = LdPk<InT>::load(A + (size_t)(m0 + r) * K + k0 + cc);
      *(uint2*)(wS + r * LDA + cc) = LdPk<u16>::load(Wb + (size_t)(n0 + r) * K + k0 + cc);
    }
    __syncthreads();

    Frag a[2];
#pragma unroll
    for (int mt = 0; mt < 2; ++mt) {
      const u16* base = aS + (wm * 32 + mt * 16 + ln) * LDA;
      a[mt].u.lo = *(const uint4*)(base + half * 8);        // k = 8h .. 8h+7
      a[mt].u.hi = *(const uint4*)(base + 16 + half * 8);   // k = 16+8h ..
    }
#pragma unroll
    for (int nt = 0; nt < 4; ++nt) {
      const u16* base = wS + (wn * 64 + nt * 16 + ln) * LDA;
      Frag bf;
      bf.u.lo = *(const uint4*)(base + half * 8);           // k = 8h .. 8h+7
      bf.u.hi = *(const uint4*)(base + 16 + half * 8);      // k = 16+8h ..
#pragma unroll
      for (int mt = 0; mt < 2; ++mt)
        c[mt][nt] = wmma_bf16(a[mt].v, bf.v, c[mt][nt]);
    }
  }

  // epilogue: C-layout row = r + 8*half, col = ln
#pragma unroll
  for (int mt = 0; mt < 2; ++mt)
#pragma unroll
    for (int nt = 0; nt < 4; ++nt) {
      int n = n0 + wn * 64 + nt * 16 + ln;
      float bv = bias[n];
#pragma unroll
      for (int r = 0; r < 8; ++r) {
        int m = m0 + wm * 32 + mt * 16 + r + 8 * half;
        float val = (c[mt][nt][r] + bv) * outScale;
        if constexpr (LAYOUT == 1) {
          // token m = (b, s); store at [(b*Dd + n)*Ss + s]
          stout<OutT>(C + ((size_t)((m >> 11) * Dd + n)) * Ss + (m & (Ss - 1)), val);
        } else if constexpr (LAYOUT == 2) {
          // permute rows within 32-token groups: s' = (s&1)<<4 | s>>1
          int mp = (m & ~31) | (((m & 1) << 4) | ((m & 31) >> 1));
          stout<OutT>(C + (size_t)mp * N + n, val);
        } else {
          stout<OutT>(C + (size_t)m * N + n, val);
        }
      }
    }
}

// ---------------------------------------------------------------------------
// Flash attention: grid (B*H, S/64), block 128 (4 waves; wave = 16 q rows).
// K/V tiles staged by TDM (wave 0 issues; TENSORcnt + barrier publishes).
// Kp rows are pre-permuted so LDS row j holds key 2j (j<16) / 2(j-16)+1.
// ---------------------------------------------------------------------------
__global__ __launch_bounds__(128)
__attribute__((amdgpu_waves_per_eu(1)))
void attn_kernel(
    const u16* __restrict__ Qp, const u16* __restrict__ Kp,
    const u16* __restrict__ Vt, u16* __restrict__ ctx) {
  __shared__ u16 kt[2][32 * 72];       // K tile [slot][d],  row 64+8 pad u16
  __shared__ u16 vt[2][64 * 40];       // V^T tile [d][kv],  row 32+8 pad u16
  __shared__ u16 pt[4 * 16 * 48];      // per-wave P tile [m][kv], stride 48

  const int bh   = blockIdx.x;
  const int b    = bh / Hh, h = bh % Hh;
  const int w    = threadIdx.x >> 5;
  const int lane = threadIdx.x & 31;
  const int half = lane >> 4, ln = lane & 15;
  const int q0   = blockIdx.y * 64 + w * 16;

  const size_t kBase = (size_t)(b * Ss) * Dd + h * DEPTH;        // K rows
  const size_t vBase = ((size_t)b * Dd + h * DEPTH) * Ss;        // Vt rows

  // Q fragments for this wave's 16 rows (held for the whole kv loop)
  Frag qa[2];
  {
    const u16* qrow = Qp + ((size_t)(b * Ss + q0 + ln)) * Dd + h * DEPTH;
#pragma unroll
    for (int dc = 0; dc < 2; ++dc) {
      qa[dc].u.lo = *(const uint4*)(qrow + dc * 32 + half * 8);
      qa[dc].u.hi = *(const uint4*)(qrow + dc * 32 + 16 + half * 8);
    }
  }

  // all-ones B fragment for the softmax-denominator WMMA
  Frag ones;
#pragma unroll
  for (int i = 0; i < 4; ++i) {
    (&ones.u.lo.x)[i] = 0x3f803f80u;
    (&ones.u.hi.x)[i] = 0x3f803f80u;
  }

  v8f o[4], osum;
#pragma unroll
  for (int df = 0; df < 4; ++df)
#pragma unroll
    for (int r = 0; r < 8; ++r) o[df][r] = 0.0f;
#pragma unroll
  for (int r = 0; r < 8; ++r) osum[r] = 0.0f;
  float rm[8];
#pragma unroll
  for (int r = 0; r < 8; ++r) rm[r] = -1e30f;

  constexpr int NT = Ss / 32;              // 64 kv tiles

  // prologue: TDM-stage tile 0 into buffer 0 (K: 32x64, V^T: 64x32)
  if (w == 0) {
    tdm_tile_to_lds<64, 32, Dd, 4, 3>(Kp + kBase,
                                      (u32)(__UINTPTR_TYPE__)&kt[0][0]);
    tdm_tile_to_lds<32, 64, Ss, 3, 3>(Vt + vBase,
                                      (u32)(__UINTPTR_TYPE__)&vt[0][0]);
  }

  for (int it = 0; it < NT; ++it) {
    const int buf = it & 1;
    if (w == 0) __builtin_amdgcn_s_wait_tensorcnt(0);  // tile `it` landed
    __syncthreads();
    const bool more = (it + 1) < NT;
    if (w == 0 && more) {                  // TDM-prefetch tile it+1
      const int kv1 = (it + 1) * 32;
      tdm_tile_to_lds<64, 32, Dd, 4, 3>(Kp + kBase + (size_t)kv1 * Dd,
                                        (u32)(__UINTPTR_TYPE__)&kt[buf ^ 1][0]);
      tdm_tile_to_lds<32, 64, Ss, 3, 3>(Vt + vBase + kv1,
                                        (u32)(__UINTPTR_TYPE__)&vt[buf ^ 1][0]);
    }

    // ---- logits: c0 cols = keys 2n, c1 cols = keys 2n+1 (rows permuted
    //      at projection time); 2 WMMAs each over the 64 head dims ----
    v8f c0, c1;
#pragma unroll
    for (int r = 0; r < 8; ++r) { c0[r] = 0.0f; c1[r] = 0.0f; }
#pragma unroll
    for (int dc = 0; dc < 2; ++dc) {
      Frag kb;
      const u16* base0 = kt[buf] + ln * 72 + dc * 32 + half * 8;
      kb.u.lo = *(const uint4*)(base0);
      kb.u.hi = *(const uint4*)(base0 + 16);
      c0 = wmma_bf16(qa[dc].v, kb.v, c0);
      const u16* base1 = kt[buf] + (16 + ln) * 72 + dc * 32 + half * 8;
      kb.u.lo = *(const uint4*)(base1);
      kb.u.hi = *(const uint4*)(base1 + 16);
      c1 = wmma_bf16(qa[dc].v, kb.v, c1);
    }

    // ---- online softmax (scores already in log2 domain, pre-scaled) ----
    u16* pw = pt + w * 16 * 48;
#pragma unroll
    for (int r = 0; r < 8; ++r) {
      float t = vmax(c0[r], c1[r]);
      t = vmax(t, swzx<SWZ_X1>(t));
      t = vmax(t, swzx<SWZ_X2>(t));
      t = vmax(t, swzx<SWZ_X4>(t));
      t = vmax(t, swzx<SWZ_X8>(t));
      float mnew = vmax(rm[r], t);
      float sc = __builtin_amdgcn_exp2f(rm[r] - mnew);
      float p0 = __builtin_amdgcn_exp2f(c0[r] - mnew);
      float p1 = __builtin_amdgcn_exp2f(c1[r] - mnew);
      rm[r] = mnew;
      o[0][r] *= sc; o[1][r] *= sc; o[2][r] *= sc; o[3][r] *= sc;
      osum[r] *= sc;
      // keys (2ln, 2ln+1) pack into the single dword at position ln
      *((u32*)(pw + (r + 8 * half) * 48) + ln) = cvt_pk_bf16(p0, p1);
    }

    // ---- reload P in A-layout (wave-local LDS round trip, plain order) ----
    Frag pa;
    {
      const u16* base = pw + ln * 48;
      pa.u.lo = *(const uint4*)(base + half * 8);
      pa.u.hi = *(const uint4*)(base + 16 + half * 8);
    }

    // ---- denominator += P * ones ; O += P * V (plain contiguous reads) ----
    osum = wmma_bf16(pa.v, ones.v, osum);
#pragma unroll
    for (int df = 0; df < 4; ++df) {
      Frag vb;
      const u16* base = vt[buf] + (df * 16 + ln) * 40 + half * 8;
      vb.u.lo = *(const uint4*)(base);
      vb.u.hi = *(const uint4*)(base + 16);
      o[df] = wmma_bf16(pa.v, vb.v, o[df]);
    }
  }

  // ---- epilogue: ctx[b, q, h*64 + d] = O / rowsum (bf16) ----
  float inv[8];
#pragma unroll
  for (int r = 0; r < 8; ++r) inv[r] = __builtin_amdgcn_rcpf(osum[r]);
#pragma unroll
  for (int df = 0; df < 4; ++df)
#pragma unroll
    for (int r = 0; r < 8; ++r) {
      int row = q0 + r + 8 * half;
      int col = h * DEPTH + df * 16 + ln;
      ctx[((size_t)(b * Ss + row)) * Dd + col] = f2bf(o[df][r] * inv[r]);
    }
}

// ---------------------------------------------------------------------------
extern "C" void kernel_launch(void* const* d_in, const int* in_sizes, int n_in,
                              void* d_out, int out_size, void* d_ws,
                              size_t ws_size, hipStream_t stream) {
  (void)in_sizes; (void)n_in; (void)out_size; (void)ws_size;
  const float* q  = (const float*)d_in[0];
  const float* k  = (const float*)d_in[1];
  const float* v  = (const float*)d_in[2];
  const float* wq = (const float*)d_in[3];
  const float* bq = (const float*)d_in[4];
  const float* wk = (const float*)d_in[5];
  const float* bk = (const float*)d_in[6];
  const float* wv = (const float*)d_in[7];
  const float* bv = (const float*)d_in[8];
  const float* wo = (const float*)d_in[9];
  const float* bo = (const float*)d_in[10];
  float* out = (float*)d_out;

  const size_t nTok = (size_t)Bb * Ss;          // 4096
  const size_t nEl  = nTok * Dd;                // 4M elements
  const size_t nW   = (size_t)Dd * Dd;          // 1M elements per weight
  u16* Qp  = (u16*)d_ws;
  u16* Kp  = Qp + nEl;                          // row-permuted within 32-groups
  u16* Vt  = Kp + nEl;                          // transposed [b, d, s]
  u16* ctx = Vt + nEl;
  u16* wqB = ctx + nEl;                         // bf16 weights
  u16* wkB = wqB + nW;
  u16* wvB = wkB + nW;
  u16* woB = wvB + nW;

  // one-shot weight conversion fp32 -> bf16
  const int nQuads = (int)(nW / 4);             // 262144
  dim3 gc((nQuads + 255) / 256);
  cvt_bf16_kernel<<<gc, 256, 0, stream>>>(wq, wqB, nQuads);
  cvt_bf16_kernel<<<gc, 256, 0, stream>>>(wk, wkB, nQuads);
  cvt_bf16_kernel<<<gc, 256, 0, stream>>>(wv, wvB, nQuads);
  cvt_bf16_kernel<<<gc, 256, 0, stream>>>(wo, woB, nQuads);

  const int M = (int)nTok, N = Dd, K = Dd;
  // log2(e)/sqrt(DEPTH): folds the attention scale AND the exp->exp2 switch
  const float qscale = 1.44269504088896340736f * 0.125f;
  dim3 gg(N / 128, M / 128);                    // (8, 32)
  gemm_bias_bf16<float, u16, 0><<<gg, 256, 0, stream>>>(q, wqB, bq, Qp, M, N, K, qscale);
  gemm_bias_bf16<float, u16, 2><<<gg, 256, 0, stream>>>(k, wkB, bk, Kp, M, N, K, 1.0f);
  gemm_bias_bf16<float, u16, 1><<<gg, 256, 0, stream>>>(v, wvB, bv, Vt, M, N, K, 1.0f);

  dim3 ga(Bb * Hh, Ss / 64);                    // (32, 32)
  attn_kernel<<<ga, 128, 0, stream>>>(Qp, Kp, Vt, ctx);

  gemm_bias_bf16<u16, float, 0><<<gg, 256, 0, stream>>>(ctx, woB, bo, out, M, N, K, 1.0f);
}
